// LSTM_15470472200790
// MI455X (gfx1250) — matrix operand order repaired and checked by reading, not compiled, not run on hardware
//
#include <hip/hip_runtime.h>

// ---------------------------------------------------------------------------
// Time-varying-weight LSTM for MI455X (gfx1250), persistent-kernel design.
//
// Memory-bound problem: 600 MB of once-used fp32 weights -> ~26 us floor at
// 23.3 TB/s. Weights are streamed fp32 -> converted in-register to f16 ->
// v_wmma_f32_16x16x32_f16 with fp32 accumulation. Activations stay in L2/regs.
// One persistent kernel walks the 96 sequential timesteps with a device-wide
// atomic generation barrier (2 syncs per step).
// ---------------------------------------------------------------------------

#define T_STEPS 96
#define BATCH   32
#define DIN_    256
#define HID     512
#define VOC     128

#define NGATE_WG 128                 // 4 gates x 32 column strips of 16
#define NLIN_WG  8                   // 128 vocab cols / 16
#define NBLK     (NGATE_WG + NLIN_WG)
#define NTHR     256                 // 8 wave32 per workgroup
#define BH       (BATCH * HID)

typedef __attribute__((ext_vector_type(16))) _Float16 v16h;
typedef __attribute__((ext_vector_type(8)))  float    v8f;

// Load 8+8 consecutive fp32 from two 16B-aligned addresses, convert to f16.
// Element e<8 -> p0[e], e>=8 -> p1[e-8]  (matches ISA 16-bit A/B VGPR packing:
// vector elem 2p / 2p+1 = lo/hi half of VGPR p).
__device__ __forceinline__ v16h cvt16(const float* __restrict__ p0,
                                      const float* __restrict__ p1) {
  const float4 a = *(const float4*)(p0);
  const float4 b = *(const float4*)(p0 + 4);
  const float4 c = *(const float4*)(p1);
  const float4 d = *(const float4*)(p1 + 4);
  v16h r;
  r[0]  = (_Float16)a.x; r[1]  = (_Float16)a.y; r[2]  = (_Float16)a.z; r[3]  = (_Float16)a.w;
  r[4]  = (_Float16)b.x; r[5]  = (_Float16)b.y; r[6]  = (_Float16)b.z; r[7]  = (_Float16)b.w;
  r[8]  = (_Float16)c.x; r[9]  = (_Float16)c.y; r[10] = (_Float16)c.z; r[11] = (_Float16)c.w;
  r[12] = (_Float16)d.x; r[13] = (_Float16)d.y; r[14] = (_Float16)d.z; r[15] = (_Float16)d.w;
  return r;
}

// Device-wide generation barrier. Counter is zeroed by init kernel per launch.
__device__ __forceinline__ void grid_sync(unsigned* cnt, unsigned target) {
  __threadfence();                     // make this wave's stores L2-visible
  __syncthreads();
  if (threadIdx.x == 0) {
    __hip_atomic_fetch_add(cnt, 1u, __ATOMIC_RELEASE, __HIP_MEMORY_SCOPE_AGENT);
    while (__hip_atomic_load(cnt, __ATOMIC_ACQUIRE, __HIP_MEMORY_SCOPE_AGENT) < target)
      __builtin_amdgcn_s_sleep(1);
  }
  __syncthreads();
  __threadfence();                     // per-wave acquire: invalidate L0/WGP$
}

// Output projection for timestep tt: logits = h[tt] @ lin_w[tt].T + lin_b[tt].
// One WG handles 16 vocab columns; 8 waves split K=512; LDS reduction.
__device__ __forceinline__ void lin_task(int tt, int v0,
    const float* __restrict__ hbuf, const float* __restrict__ lin_w,
    const float* __restrict__ lin_b, float* __restrict__ out,
    float* __restrict__ red) {
  const int lane = threadIdx.x & 31;
  const int wv   = threadIdx.x >> 5;
  const int half = lane >> 4;
  const int lr   = lane & 15;
  const int v    = v0 + lr;

  v8f acc0 = {}; v8f acc1 = {};
  const float* lrow = lin_w + ((size_t)tt * VOC + v) * HID;
  const int kb = wv * (HID / 8);                       // 64 K per wave
  for (int kk = kb; kk < kb + HID / 8; kk += 32) {
    const v16h bm = cvt16(lrow + kk + half * 8, lrow + kk + 16 + half * 8);
    const float* ar0 = hbuf + (size_t)lr * HID + kk;
    const float* ar1 = hbuf + (size_t)(16 + lr) * HID + kk;
    const v16h a0 = cvt16(ar0 + half * 8, ar0 + 16 + half * 8);
    const v16h a1 = cvt16(ar1 + half * 8, ar1 + 16 + half * 8);
    acc0 = __builtin_amdgcn_wmma_f32_16x16x32_f16(false, a0, false, bm, (short)0, acc0, false, false);
    acc1 = __builtin_amdgcn_wmma_f32_16x16x32_f16(false, a1, false, bm, (short)0, acc1, false, false);
  }
#pragma unroll
  for (int r = 0; r < 8; ++r) {
    red[wv * 512 +       r * 32 + lane] = acc0[r];
    red[wv * 512 + 256 + r * 32 + lane] = acc1[r];
  }
  __syncthreads();
  for (int j = threadIdx.x; j < 512; j += NTHR) {
    float s = 0.f;
#pragma unroll
    for (int w = 0; w < 8; ++w) s += red[w * 512 + j];
    const int mt = j >> 8, r = (j >> 5) & 7, ln = j & 31;
    const int b  = mt * 16 + (ln >> 4) * 8 + r;        // D-layout: M = r + 8*half + 16*mt
    const int vv = v0 + (ln & 15);
    out[((size_t)b * T_STEPS + tt) * VOC + vv] = s + lin_b[(size_t)tt * VOC + vv];
  }
}

__global__ void __launch_bounds__(NTHR, 1) lstm_persistent(
    const float* __restrict__ x,
    const float* __restrict__ Wif,   const float* __restrict__ Wii,
    const float* __restrict__ Wic,   const float* __restrict__ Wio,
    const float* __restrict__ Whf_w, const float* __restrict__ Whf_b,
    const float* __restrict__ Whi_w, const float* __restrict__ Whi_b,
    const float* __restrict__ Whc_w, const float* __restrict__ Whc_b,
    const float* __restrict__ Who_w, const float* __restrict__ Who_b,
    const float* __restrict__ lin_w, const float* __restrict__ lin_b,
    float* __restrict__ out,
    float* __restrict__ gbuf,        // 4*B*H gate pre-activations (f,i,c,o)
    float* __restrict__ hbuf,        // B*H hidden state
    float* __restrict__ cbuf,        // B*H cell state
    unsigned* __restrict__ cnt) {
  __shared__ float red[8 * 512];     // 16 KB: per-wave 32x16 partials

  const int lane = threadIdx.x & 31;
  const int wv   = threadIdx.x >> 5;
  const int half = lane >> 4;
  const int lr   = lane & 15;
  const int blk  = blockIdx.x;

  unsigned gen = 0;
  for (int t = 0; t < T_STEPS; ++t) {
    // ---------------- Phase A: gate GEMMs (t) + output proj (t-1) ----------
    if (blk < NGATE_WG) {
      const int g  = blk & 3;                          // 0=f 1=i 2=c 3=o
      const int n0 = (blk >> 2) * 16;
      const int n  = n0 + lr;
      const float* Wxp = (g == 0) ? Wif   : (g == 1) ? Wii   : (g == 2) ? Wic   : Wio;
      const float* Whp = (g == 0) ? Whf_w : (g == 1) ? Whi_w : (g == 2) ? Whc_w : Who_w;
      const float* Bhp = (g == 0) ? Whf_b : (g == 1) ? Whi_b : (g == 2) ? Whc_b : Who_b;

      const int totK = (t == 0) ? DIN_ : (DIN_ + HID); // t=0: x-only, no bias
      const int klen = totK >> 3;                      // K slice per wave (32/96)
      const int kb   = wv * klen;

      const float* wxrow = Wxp + ((size_t)t * HID + n) * DIN_;
      const float* whrow = Whp + ((size_t)t * HID + n) * HID;

      v8f acc0 = {}; v8f acc1 = {};
      for (int kk = kb; kk < kb + klen; kk += 32) {    // each 32-chunk fully in x or h
        v16h bm, a0, a1;
        if (kk < DIN_) {
          bm = cvt16(wxrow + kk + half * 8, wxrow + kk + 16 + half * 8);
          const float* ar0 = x + ((size_t)lr        * T_STEPS + t) * DIN_ + kk;
          const float* ar1 = x + ((size_t)(16 + lr) * T_STEPS + t) * DIN_ + kk;
          a0 = cvt16(ar0 + half * 8, ar0 + 16 + half * 8);
          a1 = cvt16(ar1 + half * 8, ar1 + 16 + half * 8);
        } else {
          const int kh = kk - DIN_;
          bm = cvt16(whrow + kh + half * 8, whrow + kh + 16 + half * 8);
          const float* ar0 = hbuf + (size_t)lr        * HID + kh;
          const float* ar1 = hbuf + (size_t)(16 + lr) * HID + kh;
          a0 = cvt16(ar0 + half * 8, ar0 + 16 + half * 8);
          a1 = cvt16(ar1 + half * 8, ar1 + 16 + half * 8);
        }
        acc0 = __builtin_amdgcn_wmma_f32_16x16x32_f16(false, a0, false, bm, (short)0, acc0, false, false);
        acc1 = __builtin_amdgcn_wmma_f32_16x16x32_f16(false, a1, false, bm, (short)0, acc1, false, false);
      }
#pragma unroll
      for (int r = 0; r < 8; ++r) {
        red[wv * 512 +       r * 32 + lane] = acc0[r];
        red[wv * 512 + 256 + r * 32 + lane] = acc1[r];
      }
      __syncthreads();
      for (int j = threadIdx.x; j < 512; j += NTHR) {
        float s = 0.f;
#pragma unroll
        for (int w = 0; w < 8; ++w) s += red[w * 512 + j];
        const int mt = j >> 8, r = (j >> 5) & 7, ln = j & 31;
        const int m  = mt * 16 + (ln >> 4) * 8 + r;
        const int nn = n0 + (ln & 15);
        const float bias = (t > 0) ? Bhp[(size_t)t * HID + nn] : 0.f;
        gbuf[(size_t)g * BH + (size_t)m * HID + nn] = s + bias;
      }
    } else if (t >= 1) {
      lin_task(t - 1, (blk - NGATE_WG) * 16, hbuf, lin_w, lin_b, out, red);
    }

    ++gen;
    grid_sync(cnt, gen * NBLK);

    // ---------------- Phase B: elementwise cell update ----------------------
    {
      const int e = blk * NTHR + threadIdx.x;          // one (b,n) per lane
      if (e < BH) {
        const float F  = gbuf[e];
        const float I  = gbuf[BH + e];
        const float Cb = gbuf[2 * BH + e];
        const float O  = gbuf[3 * BH + e];
        const float fg = 1.f / (1.f + __expf(-F));
        const float ig = 1.f / (1.f + __expf(-I));
        const float cb = tanhf(Cb);
        const float og = 1.f / (1.f + __expf(-O));
        const float c  = (t == 0) ? ig * cb : fg * cbuf[e] + ig * cb;
        cbuf[e] = c;
        hbuf[e] = og * tanhf(c);
      }
    }

    ++gen;
    grid_sync(cnt, gen * NBLK);
  }

  // Tail: logits for the final timestep (h[95] just produced).
  if (blk >= NGATE_WG)
    lin_task(T_STEPS - 1, (blk - NGATE_WG) * 16, hbuf, lin_w, lin_b, out, red);
}

__global__ void init_sync(unsigned* cnt) {
  if (threadIdx.x == 0) cnt[0] = 0u;
}

extern "C" void kernel_launch(void* const* d_in, const int* in_sizes, int n_in,
                              void* d_out, int out_size, void* d_ws, size_t ws_size,
                              hipStream_t stream) {
  (void)in_sizes; (void)n_in; (void)out_size; (void)ws_size;

  const float* x     = (const float*)d_in[0];
  const float* Wif   = (const float*)d_in[1];
  const float* Wii   = (const float*)d_in[2];
  const float* Wic   = (const float*)d_in[3];
  const float* Wio   = (const float*)d_in[4];
  const float* Whf_w = (const float*)d_in[5];
  const float* Whf_b = (const float*)d_in[6];
  const float* Whi_w = (const float*)d_in[7];
  const float* Whi_b = (const float*)d_in[8];
  const float* Whc_w = (const float*)d_in[9];
  const float* Whc_b = (const float*)d_in[10];
  const float* Who_w = (const float*)d_in[11];
  const float* Who_b = (const float*)d_in[12];
  const float* lin_w = (const float*)d_in[13];
  const float* lin_b = (const float*)d_in[14];
  float* out = (float*)d_out;

  // Workspace layout (well under any plausible ws_size):
  //   [0,256):        sync counter
  //   gbuf: 4*B*H f32, hbuf: B*H f32, cbuf: B*H f32   (~384 KB total)
  char* ws = (char*)d_ws;
  unsigned* cnt = (unsigned*)ws;
  float* gbuf = (float*)(ws + 256);
  float* hbuf = gbuf + 4 * BH;
  float* cbuf = hbuf + BH;

  init_sync<<<1, 64, 0, stream>>>(cnt);
  lstm_persistent<<<NBLK, NTHR, 0, stream>>>(
      x, Wif, Wii, Wic, Wio,
      Whf_w, Whf_b, Whi_w, Whi_b, Whc_w, Whc_b, Who_w, Who_b,
      lin_w, lin_b, out, gbuf, hbuf, cbuf, cnt);
}
